// SparseVectorQuantizer_10711648436602
// MI455X (gfx1250) — matrix operand order repaired and checked by reading, not compiled, not run on hardware
//
#include <hip/hip_runtime.h>
#include <hip/hip_bf16.h>
#include <stdint.h>

typedef __attribute__((ext_vector_type(16))) __bf16 v16bf;
typedef __attribute__((ext_vector_type(8)))  float  v8f;
typedef int v4i __attribute__((vector_size(16)));

using g_v4i = __attribute__((address_space(1))) v4i*;
using l_v4i = __attribute__((address_space(3))) v4i*;

#define NUM_EMB   8192
#define EMBED     64
#define NVOX      32768
#define NBATCH    4
#define ROWS_PER_BLOCK 128
#define CHUNK_CODES 64                 // codes staged in LDS per step
#define NCHUNK    (NUM_EMB / CHUNK_CODES)

// d_out flat layout (floats):
//   [0 .. 2097151]           quantized_st (32768 x 64)
//   [2097152]                vq_loss
//   [2097153]                commitment_loss
//   [2097154 .. 2129921]     encoding_indices_st (32768)
//   [2129922]                avg_perplexity
//   [2129923]                avg_entropy
//   [2129924]                avg_unique_count
//   [2129925]                utilization_ratio
#define OUT_Q      0
#define OUT_VQ     2097152
#define OUT_COMMIT 2097153
#define OUT_IDX    2097154
#define OUT_PERP   2129922
#define OUT_ENT    2129923
#define OUT_UNIQ   2129924
#define OUT_UTIL   2129925

// workspace layout (bytes)
#define WS_CBH    0u          // bf16 hi codebook: 8192*64*2 = 1048576
#define WS_CBL    1048576u    // bf16 lo codebook: 1048576
#define WS_CNORM  2097152u    // f32 ||c||^2: 8192*4 = 32768
#define WS_COUNTS 2129920u    // int counts: 4*8192*4 = 131072
#define WS_ACC    2260992u    // f32 acc[9]: [0]=sse, [1..4]=entropy_b, [5..8]=unique_b

// ---------------------------------------------------------------------------
// Async global -> LDS copy (16 bytes), ASYNCcnt-tracked on CDNA5
// ---------------------------------------------------------------------------
#if defined(__HIP_DEVICE_COMPILE__) && __has_builtin(__builtin_amdgcn_global_load_async_to_lds_b128)
#define HAS_ASYNC_LDS 1
#else
#define HAS_ASYNC_LDS 0
#endif

__device__ __forceinline__ void cp16_async(const void* gsrc, void* ldst) {
#if HAS_ASYNC_LDS
    __builtin_amdgcn_global_load_async_to_lds_b128((g_v4i)gsrc, (l_v4i)ldst, 0, 0);
#else
    *(float4*)ldst = *(const float4*)gsrc;  // synchronous fallback
#endif
}

#if HAS_ASYNC_LDS
#define ASYNC_WAIT(N) asm volatile("s_wait_asynccnt %0" ::"n"(N) : "memory")
#else
#define ASYNC_WAIT(N)
#endif

// ---------------------------------------------------------------------------
// Zero counters / accumulators
// ---------------------------------------------------------------------------
__global__ void vq_init(int* counts, float* accf) {
    int i = blockIdx.x * blockDim.x + threadIdx.x;
    if (i < NBATCH * NUM_EMB) counts[i] = 0;
    if (i < 9) accf[i] = 0.0f;
}

// ---------------------------------------------------------------------------
// Precompute codebook bf16 hi/lo split and per-code squared norms
// ---------------------------------------------------------------------------
__global__ __launch_bounds__(64) void vq_prep(const float* __restrict__ codebook,
                                              __bf16* __restrict__ cbh,
                                              __bf16* __restrict__ cbl,
                                              float* __restrict__ cnorm) {
    const int row = blockIdx.x;
    const int tid = threadIdx.x;
    const float x = codebook[row * EMBED + tid];
    const __bf16 h = (__bf16)x;
    cbh[row * EMBED + tid] = h;
    cbl[row * EMBED + tid] = (__bf16)(x - (float)h);
    __shared__ float s[64];
    s[tid] = x * x;
    __syncthreads();
    for (int st = 32; st; st >>= 1) {
        if (tid < st) s[tid] += s[tid + st];
        __syncthreads();
    }
    if (tid == 0) cnorm[row] = s[0];
}

// ---------------------------------------------------------------------------
// Main fused kernel: async-LDS double-buffered WMMA distance GEMM + argmin
// + gather + SSE + histogram.  One wave owns a 16-row tile of z; block = 8
// waves = 128 rows; B (codebook) tiles staged once per block in LDS.
// ---------------------------------------------------------------------------
__global__ __launch_bounds__(256) void vq_main(const float* __restrict__ z,
                                               const int* __restrict__ zc,
                                               const float* __restrict__ codebook,
                                               const __bf16* __restrict__ cbh,
                                               const __bf16* __restrict__ cbl,
                                               const float* __restrict__ cnorm,
                                               int* __restrict__ counts,
                                               float* __restrict__ accf,
                                               float* __restrict__ out) {
    // [buf][hi|lo][64 codes * 64 elems] = 32 KB double buffer
    __shared__ __align__(64) __bf16 sB[2][2][CHUNK_CODES * EMBED];
    __shared__ int   sBest[ROWS_PER_BLOCK];
    __shared__ float sred[256];

    const int tid  = threadIdx.x;
    const int wid  = tid >> 5;
    const int lane = tid & 31;
    const int l    = lane & 15;
    const int half = lane >> 4;
    const int blockRow = blockIdx.x * ROWS_PER_BLOCK;
    const int arow     = blockRow + wid * 16 + l;

    // ---- Load my A row (one z row), split fp32 -> bf16 hi + bf16 lo ----
    // A 16x32 in this code's contiguous K order: lanes 0-15 hold K {0..15}; lanes 16-31 hold
    // K {16..31}.  Two K-halves (0..31 / 32..63) => two A operands.
    const float* zr = z + (size_t)arow * EMBED;
    const int kb = half * 16;
    v16bf a1h, a1l, a2h, a2l;
#pragma unroll
    for (int i = 0; i < 8; ++i) {
        float x0 = zr[kb + i];
        float x1 = zr[kb + 8 + i];
        float x2 = zr[32 + kb + i];
        float x3 = zr[32 + kb + 8 + i];
        __bf16 h0 = (__bf16)x0, h1 = (__bf16)x1, h2 = (__bf16)x2, h3 = (__bf16)x3;
        a1h[i]     = h0;  a1l[i]     = (__bf16)(x0 - (float)h0);
        a1h[8 + i] = h1;  a1l[8 + i] = (__bf16)(x1 - (float)h1);
        a2h[i]     = h2;  a2l[i]     = (__bf16)(x2 - (float)h2);
        a2h[8 + i] = h3;  a2l[8 + i] = (__bf16)(x3 - (float)h3);
    }

    float best[8];
    int   bestIdx[8];
#pragma unroll
    for (int r = 0; r < 8; ++r) { best[r] = 3.4e38f; bestIdx[r] = 0; }

    // 4 uniform b128 async copies per thread per chunk (hi 8KB + lo 8KB)
    auto issue_chunk = [&](int chunk, int buf) {
        const size_t g = (size_t)chunk * CHUNK_CODES * EMBED;
        cp16_async(cbh + g +        (size_t)tid * 8, &sB[buf][0][tid * 8]);
        cp16_async(cbh + g + 2048 + (size_t)tid * 8, &sB[buf][0][2048 + tid * 8]);
        cp16_async(cbl + g +        (size_t)tid * 8, &sB[buf][1][tid * 8]);
        cp16_async(cbl + g + 2048 + (size_t)tid * 8, &sB[buf][1][2048 + tid * 8]);
    };

    issue_chunk(0, 0);

    for (int c = 0; c < NCHUNK; ++c) {
        const int cur = c & 1;
        if (c + 1 < NCHUNK) {
            issue_chunk(c + 1, cur ^ 1);   // prefetch next chunk into other buffer
            ASYNC_WAIT(4);                 // retire chunk c (keep next 4 in flight)
        } else {
            ASYNC_WAIT(0);
        }
        __syncthreads();                    // chunk c visible to all waves

        const int chunkBase = c * CHUNK_CODES;
        float cns[4];
#pragma unroll
        for (int tt = 0; tt < 4; ++tt) cns[tt] = cnorm[chunkBase + tt * 16 + l];

#pragma unroll
        for (int tt = 0; tt < 4; ++tt) {
            // B 32x16 layout: lane n (col) holds contiguous K in [half*16, half*16+16)
            const __bf16* bh = &sB[cur][0][(tt * 16 + l) * EMBED + half * 16];
            const __bf16* bl = &sB[cur][1][(tt * 16 + l) * EMBED + half * 16];
            v16bf b1h = *(const v16bf*)(bh);
            v16bf b2h = *(const v16bf*)(bh + 32);
            v16bf b1l = *(const v16bf*)(bl);
            v16bf b2l = *(const v16bf*)(bl + 32);

            // 3-term split-bf16 dot: hi*hi + hi*lo + lo*hi (K=64 => 2 per term)
            v8f acc = {};
            acc = __builtin_amdgcn_wmma_f32_16x16x32_bf16(false, a1h, false, b1h, (short)0, acc, false, false);
            acc = __builtin_amdgcn_wmma_f32_16x16x32_bf16(false, a2h, false, b2h, (short)0, acc, false, false);
            acc = __builtin_amdgcn_wmma_f32_16x16x32_bf16(false, a1h, false, b1l, (short)0, acc, false, false);
            acc = __builtin_amdgcn_wmma_f32_16x16x32_bf16(false, a2h, false, b2l, (short)0, acc, false, false);
            acc = __builtin_amdgcn_wmma_f32_16x16x32_bf16(false, a1l, false, b1h, (short)0, acc, false, false);
            acc = __builtin_amdgcn_wmma_f32_16x16x32_bf16(false, a2l, false, b2h, (short)0, acc, false, false);

            const int code = chunkBase + tt * 16 + l;
#pragma unroll
            for (int r = 0; r < 8; ++r) {
                // d2 = ||z||^2 + cn - 2*dot ; ||z||^2 constant per row -> drop
                float score = cns[tt] - 2.0f * acc[r];
                if (score < best[r]) { best[r] = score; bestIdx[r] = code; }
            }
        }
        __syncthreads();                    // done reading buf before it is reloaded
    }

    // ---- Cross-lane argmin within 16-lane groups (C/D row layout) ----
#pragma unroll
    for (int r = 0; r < 8; ++r) {
        float s = best[r];
        int   i0 = bestIdx[r];
        for (int off = 8; off; off >>= 1) {
            float os = __shfl_xor(s, off, 16);
            int   oi = __shfl_xor(i0, off, 16);
            if (os < s || (os == s && oi < i0)) { s = os; i0 = oi; }
        }
        best[r] = s; bestIdx[r] = i0;
    }

    if (l == 0) {
#pragma unroll
        for (int r = 0; r < 8; ++r) {
            const int localRow = wid * 16 + half * 8 + r;
            const int row = blockRow + localRow;
            const int code = bestIdx[r];
            sBest[localRow] = code;
            out[OUT_IDX + row] = (float)code;
            const int b = zc[row * 4];      // z_coords[:,0]
            atomicAdd(&counts[b * NUM_EMB + code], 1);
        }
    }
    __syncthreads();

    // ---- Cooperative gather of quantized rows + SSE accumulation ----
    float sse_loc = 0.0f;
    const float4* cb4 = (const float4*)codebook;
    const float4* z4  = (const float4*)z;
    float4* q4 = (float4*)(out + OUT_Q);
    for (int i = tid; i < ROWS_PER_BLOCK * (EMBED / 4); i += 256) {
        const int lr = i >> 4;              // 16 float4 per row
        const int cc = i & 15;
        const int row = blockRow + lr;
        const int code = sBest[lr];
        float4 q  = cb4[code * 16 + cc];
        float4 zz = z4[(size_t)row * 16 + cc];
        q4[(size_t)row * 16 + cc] = q;
        float dx = zz.x - q.x, dy = zz.y - q.y, dz = zz.z - q.z, dw = zz.w - q.w;
        sse_loc += dx * dx + dy * dy + dz * dz + dw * dw;
    }
    sred[tid] = sse_loc;
    __syncthreads();
    for (int s = 128; s; s >>= 1) {
        if (tid < s) sred[tid] += sred[tid + s];
        __syncthreads();
    }
    if (tid == 0) atomicAdd(&accf[0], sred[0]);
}

// ---------------------------------------------------------------------------
// Per-batch entropy / unique-count reduction (one block per batch)
// ---------------------------------------------------------------------------
__global__ __launch_bounds__(256) void vq_stats(const int* __restrict__ counts,
                                                float* __restrict__ accf) {
    const int b = blockIdx.x;
    const int tid = threadIdx.x;
    __shared__ float sn[256], se[256], su[256];
    float n = 0.0f;
    for (int i = tid; i < NUM_EMB; i += 256) n += (float)counts[b * NUM_EMB + i];
    sn[tid] = n;
    __syncthreads();
    for (int s = 128; s; s >>= 1) { if (tid < s) sn[tid] += sn[tid + s]; __syncthreads(); }
    const float ntot = sn[0];
    float e = 0.0f, u = 0.0f;
    for (int i = tid; i < NUM_EMB; i += 256) {
        float c = (float)counts[b * NUM_EMB + i];
        if (c > 0.0f) {
            float p = c / ntot;
            e += p * logf(p + 1e-10f);
            u += 1.0f;
        }
    }
    se[tid] = e; su[tid] = u;
    __syncthreads();
    for (int s = 128; s; s >>= 1) {
        if (tid < s) { se[tid] += se[tid + s]; su[tid] += su[tid + s]; }
        __syncthreads();
    }
    if (tid == 0) { accf[1 + b] = -se[0]; accf[5 + b] = su[0]; }
}

// ---------------------------------------------------------------------------
// Scalar outputs
// ---------------------------------------------------------------------------
__global__ void vq_finalize(const float* __restrict__ accf, float* __restrict__ out) {
    if (blockIdx.x == 0 && threadIdx.x == 0) {
        const float mse = accf[0] / (float)((size_t)NVOX * EMBED);
        out[OUT_VQ]     = mse;  // vq_loss == commitment_loss numerically
        out[OUT_COMMIT] = mse;
        float ae = 0.0f, ap = 0.0f, au = 0.0f;
        for (int b = 0; b < NBATCH; ++b) {
            ae += accf[1 + b];
            ap += expf(accf[1 + b]);
            au += accf[5 + b];
        }
        ae *= 0.25f; ap *= 0.25f; au *= 0.25f;
        out[OUT_PERP] = ap;
        out[OUT_ENT]  = ae;
        out[OUT_UNIQ] = au;
        out[OUT_UTIL] = au / (float)NUM_EMB * 100.0f;
    }
}

extern "C" void kernel_launch(void* const* d_in, const int* in_sizes, int n_in,
                              void* d_out, int out_size, void* d_ws, size_t ws_size,
                              hipStream_t stream) {
    const float* z        = (const float*)d_in[0];  // (32768, 64) f32
    const int*   zc       = (const int*)d_in[1];    // (32768, 4) i32
    const float* codebook = (const float*)d_in[2];  // (8192, 64) f32
    float* out = (float*)d_out;

    unsigned char* ws = (unsigned char*)d_ws;
    __bf16* cbh  = (__bf16*)(ws + WS_CBH);
    __bf16* cbl  = (__bf16*)(ws + WS_CBL);
    float* cnorm = (float*)(ws + WS_CNORM);
    int*   cnt   = (int*)(ws + WS_COUNTS);
    float* accf  = (float*)(ws + WS_ACC);

    vq_init<<<(NBATCH * NUM_EMB + 255) / 256, 256, 0, stream>>>(cnt, accf);
    vq_prep<<<NUM_EMB, 64, 0, stream>>>(codebook, cbh, cbl, cnorm);
    vq_main<<<NVOX / ROWS_PER_BLOCK, 256, 0, stream>>>(z, zc, codebook, cbh, cbl,
                                                       cnorm, cnt, accf, out);
    vq_stats<<<NBATCH, 256, 0, stream>>>(cnt, accf);
    vq_finalize<<<1, 32, 0, stream>>>(accf, out);
}
